// SelfAttention_50216757625443
// MI455X (gfx1250) — matrix operand order repaired and checked by reading, not compiled, not run on hardware
//
#include <hip/hip_runtime.h>
#include <hip/hip_bf16.h>

typedef __attribute__((ext_vector_type(16))) __bf16 v16bf;
typedef __attribute__((ext_vector_type(8)))  __bf16 v8bf;
typedef __attribute__((ext_vector_type(8)))  float  v8f;
typedef __attribute__((ext_vector_type(4)))  unsigned int u32x4;
typedef __attribute__((ext_vector_type(8)))  int i32x8;
typedef __attribute__((ext_vector_type(4)))  int i32x4;

// Problem constants
#define N_B   2
#define L_SEQ 2048
#define D_MOD 2048
#define H_N   16
#define HD_N  128
#define LDA_T 40   // padded LDS row stride (elements) for 32-wide tiles

__device__ __forceinline__ v8f zero8() {
    v8f z = {0.f, 0.f, 0.f, 0.f, 0.f, 0.f, 0.f, 0.f};
    return z;
}

__device__ __forceinline__ v16bf cat8(v8bf lo, v8bf hi) {
    return __builtin_shufflevector(lo, hi, 0, 1, 2, 3, 4, 5, 6, 7,
                                            8, 9, 10, 11, 12, 13, 14, 15);
}

__device__ __forceinline__ v8f wmma_bf16(v16bf a, v16bf b, v8f c) {
    return __builtin_amdgcn_wmma_f32_16x16x32_bf16(false, a, false, b,
                                                   (short)0, c, false, false);
}

// A-fragment (16x32 bf16) from LDS tile with row stride LDA_T.
// Lane l covers row (l&15); lanes>=16 take K offset +16; 16 contiguous K values.
__device__ __forceinline__ v16bf afrag_lds(const __bf16* tile, int mrel, int lane) {
    int row = mrel + (lane & 15);
    int ks  = (lane >> 4) ? 16 : 0;
    const __bf16* base = tile + row * LDA_T;
    return cat8(*(const v8bf*)(base + ks), *(const v8bf*)(base + ks + 8));
}

// B-fragment (32x16 bf16) from LDS tile storing B^T rows (row = output col).
// Lane l covers col (l&15); lanes>=16 take K offset +16; 16 contiguous K values.
__device__ __forceinline__ v16bf bfrag_lds(const __bf16* tile, int nrel, int lane) {
    int row = nrel + (lane & 15);
    int ks  = (lane >> 4) ? 16 : 0;
    const __bf16* base = tile + row * LDA_T + ks;
    return cat8(*(const v8bf*)(base), *(const v8bf*)(base + 8));
}

__device__ __forceinline__ float rowmax16(float v) {
    v = fmaxf(v, __shfl_xor(v, 1));
    v = fmaxf(v, __shfl_xor(v, 2));
    v = fmaxf(v, __shfl_xor(v, 4));
    v = fmaxf(v, __shfl_xor(v, 8));
    return v;
}

__device__ __forceinline__ float rowsum16(float v) {
    v += __shfl_xor(v, 1);
    v += __shfl_xor(v, 2);
    v += __shfl_xor(v, 4);
    v += __shfl_xor(v, 8);
    return v;
}

// ---------------------------------------------------------------------------
// Kernel A: qkv = x @ w_qkv^T (fp32 -> bf16 WMMA), fused RoPE on q,k epilogue.
// Block = 128x128 C tile = exactly one (part, head). 8 waves of 64x32.
// Outputs: q_ro, k_ro as [N,H,L,HD] bf16 (RoPE applied); v_t as [N,H,HD,L] bf16.
// ---------------------------------------------------------------------------
__global__ __launch_bounds__(256) void qkv_rope_kernel(
    const float* __restrict__ x, const float* __restrict__ pos,
    const float* __restrict__ w, __bf16* __restrict__ q_ro,
    __bf16* __restrict__ k_ro, __bf16* __restrict__ v_t)
{
    __shared__ __align__(16) unsigned char smem[32768];
    __bf16* As = (__bf16*)smem;              // 128 x LDA_T (10240 B)
    __bf16* Bs = As + 128 * LDA_T;           // 128 x LDA_T (10240 B)
    float*  Cs = (float*)smem;               // 64 x 128 fp32 (32768 B), post-loop union

    const int t    = threadIdx.x;
    const int lane = t & 31;
    const int wv   = t >> 5;
    const int hl   = lane & 15;
    const int half = lane >> 4;
    const int wm   = wv >> 2;   // 0..1 : row half of block
    const int wn   = wv & 3;    // 0..3 : 32-col strip

    const int col_base = blockIdx.x * 128;      // 0..6143
    const int row_base = blockIdx.y * 128;      // 0..4095
    const int part = col_base >> 11;            // 0=q 1=k 2=v
    const int head = (col_base & 2047) >> 7;    // head id

    v8f acc[4][2];
#pragma unroll
    for (int mt = 0; mt < 4; ++mt)
#pragma unroll
        for (int nt = 0; nt < 2; ++nt) acc[mt][nt] = zero8();

    for (int kk = 0; kk < D_MOD; kk += 32) {
        // Cooperative fp32 -> bf16 tile staging (A: x rows, B: w rows).
#pragma unroll
        for (int i = 0; i < 4; ++i) {
            int idx = t + i * 256;          // float4 index 0..1023
            int e = idx << 2;               // element 0..4095
            int r = e >> 5, c = e & 31;
            float4 fa = *(const float4*)(x + (size_t)(row_base + r) * D_MOD + kk + c);
            __bf16* da = As + r * LDA_T + c;
            da[0] = (__bf16)fa.x; da[1] = (__bf16)fa.y;
            da[2] = (__bf16)fa.z; da[3] = (__bf16)fa.w;
            float4 fb = *(const float4*)(w + (size_t)(col_base + r) * D_MOD + kk + c);
            __bf16* db = Bs + r * LDA_T + c;
            db[0] = (__bf16)fb.x; db[1] = (__bf16)fb.y;
            db[2] = (__bf16)fb.z; db[3] = (__bf16)fb.w;
        }
        if (kk + 32 < D_MOD) {   // prefetch next K-tile (global_prefetch_b8)
            __builtin_prefetch(x + (size_t)(row_base + (t >> 1)) * D_MOD + kk + 32 + (t & 1) * 16, 0, 1);
            __builtin_prefetch(w + (size_t)(col_base + (t >> 1)) * D_MOD + kk + 32 + (t & 1) * 16, 0, 1);
        }
        __syncthreads();

        v16bf bfr0 = bfrag_lds(Bs, wn * 32, lane);
        v16bf bfr1 = bfrag_lds(Bs, wn * 32 + 16, lane);
#pragma unroll
        for (int mt = 0; mt < 4; ++mt) {
            v16bf af = afrag_lds(As, wm * 64 + mt * 16, lane);
            acc[mt][0] = wmma_bf16(af, bfr0, acc[mt][0]);
            acc[mt][1] = wmma_bf16(af, bfr1, acc[mt][1]);
        }
        __syncthreads();
    }

    if (part == 2) {
        // V: store transposed [N,H,HD,L]
#pragma unroll
        for (int mt = 0; mt < 4; ++mt)
#pragma unroll
            for (int nt = 0; nt < 2; ++nt)
#pragma unroll
                for (int r = 0; r < 8; ++r) {
                    int rb = wm * 64 + mt * 16 + half * 8 + r;
                    int cb = wn * 32 + nt * 16 + hl;        // == hd (block covers one head)
                    int mg = row_base + rb;
                    int nn = mg >> 11, l = mg & (L_SEQ - 1);
                    v_t[(((size_t)nn * H_N + head) * HD_N + cb) * L_SEQ + l] =
                        (__bf16)acc[mt][nt][r];
                }
    } else {
        __bf16* dst = (part == 0) ? q_ro : k_ro;
        for (int rh = 0; rh < 2; ++rh) {     // two 64-row phases through LDS
            __syncthreads();
            if (wm == rh) {
#pragma unroll
                for (int mt = 0; mt < 4; ++mt)
#pragma unroll
                    for (int nt = 0; nt < 2; ++nt)
#pragma unroll
                        for (int r = 0; r < 8; ++r)
                            Cs[(mt * 16 + half * 8 + r) * 128 + wn * 32 + nt * 16 + hl] =
                                acc[mt][nt][r];
            }
            __syncthreads();
#pragma unroll
            for (int i = 0; i < 32; ++i) {
                int idx = i * 256 + t;
                int r64 = idx >> 7, c = idx & 127;          // c == hd
                int mg = row_base + rh * 64 + r64;
                int nn = mg >> 11, l = mg & (L_SEQ - 1);
                float val = Cs[r64 * 128 + c];
                float prt = Cs[r64 * 128 + (c ^ 64)];
                float rot = (c < 64) ? -prt : prt;          // rotate_half
                float co = pos[l * HD_N + c];
                float si = pos[L_SEQ * HD_N + l * HD_N + c];
                dst[(((size_t)nn * H_N + head) * L_SEQ + l) * HD_N + c] =
                    (__bf16)(val * co + rot * si);
            }
        }
    }
}

// ---------------------------------------------------------------------------
// Kernel B: causal flash attention. Block = (n, h, 128-row q-tile), 8 waves.
// Wave owns 16 q-rows; online softmax; S=Q K^T and O=P V via WMMA bf16.
// ---------------------------------------------------------------------------
__global__ __launch_bounds__(256) void attn_kernel(
    const __bf16* __restrict__ q_ro, const __bf16* __restrict__ k_ro,
    const __bf16* __restrict__ v_t, __bf16* __restrict__ o_b)
{
    __shared__ __align__(16) __bf16 Ps[8 * 16 * LDA_T];   // per-wave 16x32 P tiles

    const int t    = threadIdx.x;
    const int lane = t & 31;
    const int wv   = t >> 5;
    const int hl   = lane & 15;
    const int half = lane >> 4;
    const int qt   = blockIdx.x;
    const int h    = blockIdx.y;
    const int n    = blockIdx.z;

    const size_t qk_base = (size_t)(n * H_N + h) * L_SEQ * HD_N;
    const int qrow0 = qt * 128 + wv * 16;

    // Q A-fragments over HD (4 chunks of 32), loaded once
    v16bf aq[4];
    {
        const __bf16* qp = q_ro + qk_base + (size_t)(qrow0 + hl) * HD_N;
        int ks = half ? 16 : 0;
#pragma unroll
        for (int kc = 0; kc < 4; ++kc)
            aq[kc] = cat8(*(const v8bf*)(qp + kc * 32 + ks),
                          *(const v8bf*)(qp + kc * 32 + ks + 8));
    }

    v8f oa[8];
#pragma unroll
    for (int nb = 0; nb < 8; ++nb) oa[nb] = zero8();
    float mrow[8], lrow[8];
#pragma unroll
    for (int r = 0; r < 8; ++r) { mrow[r] = -3.0e38f; lrow[r] = 0.f; }

    const float scale = 0.08838834764831845f;   // 1/sqrt(128)
    const int nkb = (qt + 1) * 4;               // causal: kv blocks of 32

    for (int kb = 0; kb < nkb; ++kb) {
        const int kv0 = kb * 32;

        // S = Q K^T for two 16-col sub-tiles
        v8f st[2];
        st[0] = zero8(); st[1] = zero8();
#pragma unroll
        for (int tt = 0; tt < 2; ++tt) {
            const __bf16* kp = k_ro + qk_base + (size_t)(kv0 + tt * 16 + hl) * HD_N;
            int ks = half ? 16 : 0;
#pragma unroll
            for (int kc = 0; kc < 4; ++kc) {
                v16bf bk = *(const v16bf*)(kp + kc * 32 + ks);
                st[tt] = wmma_bf16(aq[kc], bk, st[tt]);
            }
        }

        // Online softmax update (row = half*8 + r, cols across 16-lane halves)
        float fscale[8];
#pragma unroll
        for (int r = 0; r < 8; ++r) {
            int row = qrow0 + half * 8 + r;
            float s0 = st[0][r] * scale;
            float s1 = st[1][r] * scale;
            if (kv0 + hl > row)      s0 = -3.0e38f;
            if (kv0 + 16 + hl > row) s1 = -3.0e38f;
            float tm = rowmax16(fmaxf(s0, s1));
            float mn = fmaxf(mrow[r], tm);
            float f  = __expf(mrow[r] - mn);
            float p0 = __expf(s0 - mn);
            float p1 = __expf(s1 - mn);
            lrow[r] = lrow[r] * f + rowsum16(p0 + p1);
            mrow[r] = mn;
            fscale[r] = f;
            st[0][r] = p0; st[1][r] = p1;
        }
#pragma unroll
        for (int nb = 0; nb < 8; ++nb)
#pragma unroll
            for (int r = 0; r < 8; ++r) oa[nb][r] *= fscale[r];

        // P (C-layout) -> LDS -> A-fragment layout
        __bf16* pw = Ps + wv * 16 * LDA_T;
#pragma unroll
        for (int r = 0; r < 8; ++r) {
            pw[(half * 8 + r) * LDA_T + hl]      = (__bf16)st[0][r];
            pw[(half * 8 + r) * LDA_T + 16 + hl] = (__bf16)st[1][r];
        }
        __syncthreads();
        v16bf ap;
        {
            int ks = half ? 16 : 0;
            const __bf16* pr = pw + hl * LDA_T;
            ap = cat8(*(const v8bf*)(pr + ks), *(const v8bf*)(pr + ks + 8));
        }

        // O += P V : V^T layout gives contiguous 32B B-fragment loads
#pragma unroll
        for (int nb = 0; nb < 8; ++nb) {
            const __bf16* vp = v_t + ((size_t)(n * H_N + h) * HD_N + nb * 16 + hl) * L_SEQ
                             + kv0 + (half ? 16 : 0);
            v16bf bv = *(const v16bf*)vp;
            oa[nb] = wmma_bf16(ap, bv, oa[nb]);
        }
        __syncthreads();
    }

    // Normalize and emit o as bf16 [N,L,D]
#pragma unroll
    for (int nb = 0; nb < 8; ++nb)
#pragma unroll
        for (int r = 0; r < 8; ++r) {
            int row = qrow0 + half * 8 + r;
            int hd  = nb * 16 + hl;
            o_b[((size_t)n * L_SEQ + row) * D_MOD + h * HD_N + hd] =
                (__bf16)(oa[nb][r] / lrow[r]);
        }
}

// ---------------------------------------------------------------------------
// Kernel C: out = o @ w_o^T, fp32 output. A-tile (bf16 in memory) is moved by
// the Tensor Data Mover (tensor_load_to_lds) with LDS padding reproducing the
// stride-40 tile layout; B-tile staged manually (fp32->bf16 conversion).
// ---------------------------------------------------------------------------
__global__ __launch_bounds__(256) void oproj_kernel(
    const __bf16* __restrict__ o_b, const float* __restrict__ w_o,
    float* __restrict__ out)
{
    __shared__ __align__(16) unsigned char smem[20480];
    __bf16* As = (__bf16*)smem;
    __bf16* Bs = As + 128 * LDA_T;

    const int t    = threadIdx.x;
    const int lane = t & 31;
    const int wv   = t >> 5;
    const int hl   = lane & 15;
    const int half = lane >> 4;
    const int wm   = wv >> 2;
    const int wn   = wv & 3;

    const int col_base = blockIdx.x * 128;   // 0..2047
    const int row_base = blockIdx.y * 128;   // 0..4095

    v8f acc[4][2];
#pragma unroll
    for (int mt = 0; mt < 4; ++mt)
#pragma unroll
        for (int nt = 0; nt < 2; ++nt) acc[mt][nt] = zero8();

    // TDM D# invariants for the A tile: 128x32 bf16 tile of a 4096x2048 tensor.
    const unsigned lds_a = (unsigned)(size_t)(const void*)As;   // LDS byte address
    const unsigned long long ga_base =
        (unsigned long long)(size_t)(const void*)(o_b + (size_t)row_base * D_MOD);

    for (int kk = 0; kk < D_MOD; kk += 32) {
        if (wv == 0) {
            unsigned long long ga = ga_base + (unsigned long long)kk * 2ull;
            u32x4 g0;
            g0[0] = 1u;                                   // count=1, no gather
            g0[1] = lds_a;                                // lds_addr (bytes)
            g0[2] = (unsigned)ga;                         // global_addr[31:0]
            g0[3] = (unsigned)((ga >> 32) & 0x1FFFFFFull) // global_addr[56:32]
                  | (2u << 30);                           // type=2 ("image")
            i32x8 g1;
            g1[0] = (int)((1u << 16)      // data_size = 2 bytes
                        | (1u << 20)      // pad_enable
                        | (3u << 22)      // pad_interval: 16 DWORDs (64B row)
                        | (3u << 25));    // pad_amount: 4 DWORDs (16B pad)
            g1[1] = (int)(2048u << 16);   // tensor_dim0 = 2048 (bits 79:48)
            g1[2] = (int)(4096u << 16);   // tensor_dim1 = 4096 (bits 111:80)
            g1[3] = (int)(32u << 16);     // tile_dim0 = 32 (bits 127:112)
            g1[4] = 128;                  // tile_dim1 = 128
            g1[5] = 2048;                 // tensor_dim0_stride (low 32)
            g1[6] = 0;
            g1[7] = 0;
            i32x4 gz = {0, 0, 0, 0};
#if __has_include(<hip/amd_detail/amd_gfx1250_TDM.h>)
            i32x8 gz8 = {0, 0, 0, 0, 0, 0, 0, 0};
            __builtin_amdgcn_tensor_load_to_lds(g0, g1, gz, gz, gz8, 0);
#else
            __builtin_amdgcn_tensor_load_to_lds(g0, g1, gz, gz, 0);
#endif
            __builtin_amdgcn_s_wait_tensorcnt(0);
        }
#pragma unroll
        for (int i = 0; i < 4; ++i) {   // B tile: w_o fp32 -> bf16
            int idx = t + i * 256;
            int e = idx << 2;
            int r = e >> 5, c = e & 31;
            float4 fb = *(const float4*)(w_o + (size_t)(col_base + r) * D_MOD + kk + c);
            __bf16* db = Bs + r * LDA_T + c;
            db[0] = (__bf16)fb.x; db[1] = (__bf16)fb.y;
            db[2] = (__bf16)fb.z; db[3] = (__bf16)fb.w;
        }
        if (kk + 32 < D_MOD)
            __builtin_prefetch(w_o + (size_t)(col_base + (t >> 1)) * D_MOD + kk + 32 + (t & 1) * 16, 0, 1);
        __syncthreads();

        v16bf bfr0 = bfrag_lds(Bs, wn * 32, lane);
        v16bf bfr1 = bfrag_lds(Bs, wn * 32 + 16, lane);
#pragma unroll
        for (int mt = 0; mt < 4; ++mt) {
            v16bf af = afrag_lds(As, wm * 64 + mt * 16, lane);
            acc[mt][0] = wmma_bf16(af, bfr0, acc[mt][0]);
            acc[mt][1] = wmma_bf16(af, bfr1, acc[mt][1]);
        }
        __syncthreads();
    }

#pragma unroll
    for (int mt = 0; mt < 4; ++mt)
#pragma unroll
        for (int nt = 0; nt < 2; ++nt)
#pragma unroll
            for (int r = 0; r < 8; ++r) {
                int rb = wm * 64 + mt * 16 + half * 8 + r;
                int cb = wn * 32 + nt * 16 + hl;
                out[(size_t)(row_base + rb) * D_MOD + col_base + cb] = acc[mt][nt][r];
            }
}

// ---------------------------------------------------------------------------
extern "C" void kernel_launch(void* const* d_in, const int* in_sizes, int n_in,
                              void* d_out, int out_size, void* d_ws, size_t ws_size,
                              hipStream_t stream)
{
    (void)in_sizes; (void)n_in; (void)out_size; (void)ws_size;

    const float* x     = (const float*)d_in[0];   // [N,L,D]
    const float* pos   = (const float*)d_in[1];   // [2,L,HD]
    const float* w_qkv = (const float*)d_in[2];   // [3D,D]
    const float* w_o   = (const float*)d_in[3];   // [D,D]
    float* out = (float*)d_out;                    // [N,L,D]

    const size_t SEG = (size_t)N_B * H_N * L_SEQ * HD_N;  // 8388608 elems
    __bf16* q_ro = (__bf16*)d_ws;
    __bf16* k_ro = q_ro + SEG;
    __bf16* v_t  = k_ro + SEG;
    __bf16* o_b  = v_t + SEG;   // [N,L,D] bf16

    qkv_rope_kernel<<<dim3(48, 32), 256, 0, stream>>>(x, pos, w_qkv, q_ro, k_ro, v_t);
    attn_kernel<<<dim3(16, 16, 2), 256, 0, stream>>>(q_ro, k_ro, v_t, o_b);
    oproj_kernel<<<dim3(16, 32), 256, 0, stream>>>(o_b, w_o, out);
}
